// E87SparseBlockCell_11416023073340
// MI455X (gfx1250) — matrix operand order repaired and checked by reading, not compiled, not run on hardware
//
#include <hip/hip_runtime.h>
#include <hip/hip_bf16.h>
#include <math.h>

// ---------------------------------------------------------------------------
// Problem constants (from the reference)
// ---------------------------------------------------------------------------
#define T_STEPS   512
#define BATCH     64
#define DIM       1024
#define NB        16      // N_BLOCKS
#define NS        64      // N_STATE
#define TOPK      2
#define M_ROWS    (T_STEPS * BATCH)          // 32768 GEMM rows
#define F_LOGITS  NB                         // 16
#define F_KV      (NB * 2 * NS)              // 2048
#define F_BETA    (NB * NS)                  // 1024
#define F_Q       NS                         // 64

typedef __attribute__((ext_vector_type(16))) __bf16        v16bf;
typedef __attribute__((ext_vector_type(8)))  float         v8f;
typedef __attribute__((ext_vector_type(4)))  unsigned int  u32x4;
typedef __attribute__((ext_vector_type(8)))  int           i32x8;
typedef __attribute__((ext_vector_type(4)))  int           i32x4;

#if defined(__HIP_DEVICE_COMPILE__) && __has_builtin(__builtin_amdgcn_tensor_load_to_lds) && __has_builtin(__builtin_amdgcn_s_wait_tensorcnt)
#define HAVE_TDM 1
#else
#define HAVE_TDM 0
#endif

// LDS layout for the staged W tile: 64 rows x 32 f32, TDM-padded 4 DWORDs per
// 32 -> row stride 36 floats (144 B: 16B-aligned, bank-conflict-free).
#define BROW_STRIDE 36
#define BTILE_F     (64 * BROW_STRIDE + 16)   // floats per buffer (+ trailing pad)

#if HAVE_TDM
// Issue one TDM 2D tile load: rows_valid x 32 f32 tile of W (row stride DIM)
// into LDS at byte offset lds_off. Rows >= rows_valid are zero-filled (OOB).
// Toolchain here uses the 6-arg builtin: (g0, g1, g2, g3, g4, cpol).
__device__ __forceinline__ void tdm_load_w_tile(const float* gsrc,
                                                unsigned lds_off,
                                                int rows_valid) {
    unsigned long long ga = (unsigned long long)(uintptr_t)gsrc;
    u32x4 g0;
    g0[0] = 1u;                                        // count=1, user-mode
    g0[1] = lds_off;                                   // lds_addr (bytes)
    g0[2] = (unsigned)(ga & 0xFFFFFFFFu);              // global_addr[31:0]
    g0[3] = (unsigned)((ga >> 32) & 0x01FFFFFFu)       // global_addr[56:32]
          | (2u << 30);                                // type = 2 ("image")
    i32x8 g1;
    g1[0] = (int)((2u << 16)        // data_size = 4 B
          |       (1u << 20)        // pad_enable
          |       (4u << 22)        // pad_interval: every 32 DWORDs
          |       (3u << 25));      // pad_amount: 4 DWORDs (16 B)
    g1[1] = (int)(32u << 16);                          // tensor_dim0 = 32
    g1[2] = (int)(((unsigned)rows_valid) << 16);       // tensor_dim1 = rows_valid
    g1[3] = (int)(32u << 16);                          // tile_dim0 = 32
    g1[4] = 64;                                        // tile_dim1 = 64, tile_dim2 = 0
    g1[5] = DIM;                                       // tensor_dim0_stride = 1024
    g1[6] = 0;
    g1[7] = 0;
    i32x4 z4 = {0, 0, 0, 0};
    i32x8 z8 = {0, 0, 0, 0, 0, 0, 0, 0};
    __builtin_amdgcn_tensor_load_to_lds(g0, g1, z4, z4, z8, 0);
}
#endif

// ---------------------------------------------------------------------------
// Phase 1: Y[m, n] = sum_k X[m,k] * W[n,k]   (X: M x 1024 f32, W: N x 1024 f32)
// Block = 256 thr (8 waves) computes a 128(M) x 64(N) tile. Each wave owns a
// 16x64 strip: one A fragment reused by 4 WMMAs per K-step. W tile staged in
// LDS by the Tensor Data Mover, double-buffered (TENSORcnt pipelined).
// grid = (ceil(N/64), M/128).
// ---------------------------------------------------------------------------
__global__ __launch_bounds__(256) void proj_gemm_wmma(
    const float* __restrict__ X,
    const float* __restrict__ W,
    float* __restrict__ Y,
    int Nfeat)
{
    __shared__ float Bs[2][BTILE_F];    // only LDS object -> byte offset 0

    const int lane  = threadIdx.x & 31;
    const int wave  = threadIdx.x >> 5;
    const int halfs = lane >> 4;                  // 0: lanes 0-15, 1: lanes 16-31
    const int lrow  = lane & 15;

    const int Nbase = blockIdx.x * 64;
    const int tileM = blockIdx.y * 8 + wave;

    // A-fragment addressing (16x32 bf16): lane half selects K sub-ranges.
    const int m     = tileM * 16 + lrow;
    const int koff  = halfs * 8;
    // B-fragment addressing (32x16 bf16): lane half selects K sub-ranges (same as A).
    const int khalf = halfs * 8;

    const float* __restrict__ xrow = X + (size_t)m * DIM;

    int rows_valid = Nfeat - Nbase;
    if (rows_valid > 64) rows_valid = 64;

    v8f acc[4] = {};

#if HAVE_TDM
    if (wave == 0)
        tdm_load_w_tile(W + (size_t)Nbase * DIM, 0u, rows_valid);
#endif

    for (int kt = 0; kt < 32; ++kt) {
        const int k0 = kt * 32;
        const float* __restrict__ bt = &Bs[kt & 1][0];

#if HAVE_TDM
        if (wave == 0) {
            if (kt + 1 < 32) {
                tdm_load_w_tile(W + (size_t)Nbase * DIM + (k0 + 32),
                                (unsigned)(((kt + 1) & 1) * BTILE_F * 4), rows_valid);
                __builtin_amdgcn_s_wait_tensorcnt(1);   // tile kt complete
            } else {
                __builtin_amdgcn_s_wait_tensorcnt(0);
            }
        }
        __syncthreads();
#else
        __syncthreads();
        {
            float* btw = &Bs[kt & 1][0];
            for (int idx = threadIdx.x; idx < 64 * 32; idx += 256) {
                int r = idx >> 5, cc = idx & 31;
                btw[r * BROW_STRIDE + cc] =
                    (r < rows_valid) ? W[(size_t)(Nbase + r) * DIM + k0 + cc] : 0.0f;
            }
        }
        __syncthreads();
#endif

        // ---- A fragment (loaded once, reused by 4 WMMAs)
        float af[16];
        *(float4*)(af + 0)  = *(const float4*)(xrow + k0 + koff + 0);
        *(float4*)(af + 4)  = *(const float4*)(xrow + k0 + koff + 4);
        *(float4*)(af + 8)  = *(const float4*)(xrow + k0 + koff + 16);
        *(float4*)(af + 12) = *(const float4*)(xrow + k0 + koff + 20);
        v16bf a;
#pragma unroll
        for (int i = 0; i < 16; ++i) a[i] = (__bf16)af[i];

#pragma unroll
        for (int nn = 0; nn < 4; ++nn) {
            const float* brow = bt + (nn * 16 + lrow) * BROW_STRIDE + khalf;
            float bf[16];
            *(float4*)(bf + 0)  = *(const float4*)(brow + 0);
            *(float4*)(bf + 4)  = *(const float4*)(brow + 4);
            *(float4*)(bf + 8)  = *(const float4*)(brow + 16);
            *(float4*)(bf + 12) = *(const float4*)(brow + 20);
            v16bf b;
#pragma unroll
            for (int i = 0; i < 16; ++i) b[i] = (__bf16)bf[i];

            acc[nn] = __builtin_amdgcn_wmma_f32_16x16x32_bf16(
                          false, a, false, b, (short)0, acc[nn], false, false);
        }
        __syncthreads();   // protect buffer being overwritten by next TDM
    }

    // C/D layout: VGPR r holds row M=r (lanes 0-15) / M=8+r (lanes 16-31).
    const int mbase = tileM * 16 + halfs * 8;
#pragma unroll
    for (int nn = 0; nn < 4; ++nn) {
        const int ncol = Nbase + nn * 16 + lrow;
        if (ncol < Nfeat) {
#pragma unroll
            for (int r = 0; r < 8; ++r)
                Y[(size_t)(mbase + r) * Nfeat + ncol] = acc[nn][r];
        }
    }
}

// ---------------------------------------------------------------------------
// Phase 2: sequential scan. One workgroup per batch element; state S
// (16 x 64 x 64 f32 = 256 KB) lives in dynamic LDS (WGP has 320 KB on CDNA5)
// for all 512 steps. All inner dot products / updates are float4-vectorized
// LDS traffic; next step's projection rows are prefetched into cache.
// ---------------------------------------------------------------------------
__global__ __launch_bounds__(256) void scan_kernel(
    const float* __restrict__ lg_ws,    // M x 16
    const float* __restrict__ kv_ws,    // M x 2048
    const float* __restrict__ beta_ws,  // M x 1024 (pre-bias, pre-sigmoid)
    const float* __restrict__ q_ws,     // M x 64
    const float* __restrict__ b_beta,   // 16 x 64
    float* __restrict__ out,            // T x BATCH x 64
    float* __restrict__ s_final)        // BATCH x 16 x 64 x 64
{
    extern __shared__ float smem[];
    float* S      = smem;                    // 16*64*64 = 65536
    float* q_sh   = S      + NB * NS * NS;   // 64
    float* w_sh   = q_sh   + NS;             // 16 (logits, then softmax w)
    float* kbuf   = w_sh   + NB;             // 2*64 raw k
    float* vbuf   = kbuf   + 2 * NS;         // 2*64 v
    float* bbuf   = vbuf   + 2 * NS;         // 2*64 sigmoid(beta)
    float* knb    = bbuf   + 2 * NS;         // 2*64 normalized k
    float* dbuf   = knb    + 2 * NS;         // 2*64 delta
    float* ninv   = dbuf   + 2 * NS;         // 2   inverse norms
    float* opart  = ninv   + 2;              // 4*64 partial outputs
    __shared__ int topi[TOPK];

    const int b   = blockIdx.x;
    const int tid = threadIdx.x;

    for (int idx = tid; idx < NB * NS * NS; idx += 256) S[idx] = 0.0f;
    __syncthreads();

    for (int t = 0; t < T_STEPS; ++t) {
        const size_t m = (size_t)t * BATCH + b;

        // Prefetch next step's streams while this step computes.
        if (t + 1 < T_STEPS) {
            const size_t m2 = m + BATCH;
            if (tid < 64)                __builtin_prefetch(kv_ws   + m2 * F_KV   + tid * 32, 0, 0);
            else if (tid < 96)           __builtin_prefetch(beta_ws + m2 * F_BETA + (tid - 64) * 32, 0, 0);
            else if (tid == 96)          __builtin_prefetch(q_ws    + m2 * F_Q, 0, 0);
            else if (tid == 97)          __builtin_prefetch(lg_ws   + m2 * F_LOGITS, 0, 0);
        }

        if (tid < NS) q_sh[tid] = q_ws[m * F_Q + tid];
        if (tid < NB) w_sh[tid] = lg_ws[m * F_LOGITS + tid];
        __syncthreads();

        if (tid == 0) {
            // top-2 (distinct indices, ties -> lower index, like lax.top_k)
            int i0 = 0;
            for (int c = 1; c < NB; ++c) if (w_sh[c] > w_sh[i0]) i0 = c;
            int i1 = (i0 == 0) ? 1 : 0;
            for (int c = 0; c < NB; ++c)
                if (c != i0 && w_sh[c] > w_sh[i1]) i1 = c;
            topi[0] = i0; topi[1] = i1;
            // softmax over all 16 logits
            float mx = w_sh[0];
            for (int c = 1; c < NB; ++c) mx = fmaxf(mx, w_sh[c]);
            float sum = 0.0f;
            for (int c = 0; c < NB; ++c) { w_sh[c] = expf(w_sh[c] - mx); sum += w_sh[c]; }
            float inv = 1.0f / sum;
            for (int c = 0; c < NB; ++c) w_sh[c] *= inv;
        }
        __syncthreads();

        // ---- update the two selected blocks; group g = tid>>7 (128 thr each)
        const int g  = tid >> 7;
        const int gt = tid & 127;
        const int c  = topi[g];
        if (gt < NS) {
            kbuf[g * NS + gt] = kv_ws[m * F_KV + c * (2 * NS) + gt];
            vbuf[g * NS + gt] = kv_ws[m * F_KV + c * (2 * NS) + NS + gt];
            float bz = beta_ws[m * F_BETA + c * NS + gt] + b_beta[c * NS + gt];
            bbuf[g * NS + gt] = 1.0f / (1.0f + expf(-bz));
        }
        __syncthreads();

        if (gt == 0) {
            float s2 = 0.0f;
            for (int j = 0; j < NS; ++j) s2 += kbuf[g * NS + j] * kbuf[g * NS + j];
            ninv[g] = 1.0f / (sqrtf(s2) + 1e-6f);
        }
        __syncthreads();

        if (gt < NS) knb[g * NS + gt] = kbuf[g * NS + gt] * ninv[g];
        __syncthreads();

        if (gt < NS) {   // retrieved_i = S[c] row_i . k_norm ; delta = v - retrieved
            const float4* s4 = (const float4*)(S + c * NS * NS + gt * NS);
            const float4* k4 = (const float4*)(knb + g * NS);
            float r = 0.0f;
#pragma unroll
            for (int j = 0; j < NS / 4; ++j) {
                float4 sv = s4[j], kv4 = k4[j];
                r += sv.x * kv4.x + sv.y * kv4.y + sv.z * kv4.z + sv.w * kv4.w;
            }
            dbuf[g * NS + gt] = vbuf[g * NS + gt] - r;
        }
        __syncthreads();

        // S_new = tanh(beta_i * S + delta_i * knorm_j); 2 threads per row,
        // 32 contiguous elements each (float4 LDS traffic).
        {
            const int i  = gt >> 1;
            const int jb = (gt & 1) * 32;
            const float bi = bbuf[g * NS + i];
            const float di = dbuf[g * NS + i];
            float4*       srow = (float4*)(S + c * NS * NS + i * NS + jb);
            const float4* kn   = (const float4*)(knb + g * NS + jb);
#pragma unroll
            for (int j4 = 0; j4 < 8; ++j4) {
                float4 sv = srow[j4], kv4 = kn[j4];
                sv.x = tanhf(bi * sv.x + di * kv4.x);
                sv.y = tanhf(bi * sv.y + di * kv4.y);
                sv.z = tanhf(bi * sv.z + di * kv4.z);
                sv.w = tanhf(bi * sv.w + di * kv4.w);
                srow[j4] = sv;
            }
        }
        __syncthreads();

        // ---- Sq over all 16 blocks, silu gating, weighted reduce over blocks
        {
            const int i  = tid & 63;
            const int cg = tid >> 6;    // 0..3, each covers 4 blocks
            const float4* q4 = (const float4*)q_sh;
            float acc = 0.0f;
            for (int cc = cg; cc < NB; cc += 4) {
                const float4* s4 = (const float4*)(S + cc * NS * NS + i * NS);
                float sq = 0.0f;
#pragma unroll
                for (int j = 0; j < NS / 4; ++j) {
                    float4 sv = s4[j], qv = q4[j];
                    sq += sv.x * qv.x + sv.y * qv.y + sv.z * qv.z + sv.w * qv.w;
                }
                float sig = 1.0f / (1.0f + expf(-sq));
                acc += w_sh[cc] * sq * sq * sig;    // Sq * silu(Sq)
            }
            opart[cg * NS + i] = acc;
        }
        __syncthreads();
        if (tid < NS)
            out[m * NS + tid] = opart[tid] + opart[NS + tid] +
                                opart[2 * NS + tid] + opart[3 * NS + tid];
        __syncthreads();
    }

    // S_final: (BATCH, 16, 64, 64)
    for (int idx = tid; idx < NB * NS * NS; idx += 256)
        s_final[(size_t)b * NB * NS * NS + idx] = S[idx];
}

// ---------------------------------------------------------------------------
// Host side
// ---------------------------------------------------------------------------
extern "C" void kernel_launch(void* const* d_in, const int* in_sizes, int n_in,
                              void* d_out, int out_size, void* d_ws, size_t ws_size,
                              hipStream_t stream) {
    const float* x        = (const float*)d_in[0];
    const float* W_router = (const float*)d_in[1];
    const float* W_kv     = (const float*)d_in[2];
    const float* W_beta   = (const float*)d_in[3];
    const float* b_beta   = (const float*)d_in[4];
    const float* W_q      = (const float*)d_in[5];

    float* ws     = (float*)d_ws;
    float* lg_ws  = ws;                                   // M x 16
    float* kv_ws  = lg_ws  + (size_t)M_ROWS * F_LOGITS;   // M x 2048
    float* bt_ws  = kv_ws  + (size_t)M_ROWS * F_KV;       // M x 1024
    float* q_ws   = bt_ws  + (size_t)M_ROWS * F_BETA;     // M x 64

    float* out     = (float*)d_out;                              // T x BATCH x 64
    float* s_final = out + (size_t)T_STEPS * BATCH * NS;         // BATCH x 16 x 64 x 64

    dim3 blk(256);
    proj_gemm_wmma<<<dim3((F_LOGITS + 63) / 64, M_ROWS / 128), blk, 0, stream>>>(x, W_router, lg_ws, F_LOGITS);
    proj_gemm_wmma<<<dim3((F_KV     + 63) / 64, M_ROWS / 128), blk, 0, stream>>>(x, W_kv,     kv_ws, F_KV);
    proj_gemm_wmma<<<dim3((F_BETA   + 63) / 64, M_ROWS / 128), blk, 0, stream>>>(x, W_beta,   bt_ws, F_BETA);
    proj_gemm_wmma<<<dim3((F_Q      + 63) / 64, M_ROWS / 128), blk, 0, stream>>>(x, W_q,      q_ws,  F_Q);

    // 256 KB state + small buffers of dynamic LDS (WGP has 320 KB on CDNA5)
    size_t shmem = (size_t)(NB * NS * NS + NS + NB + 5 * 2 * NS + 2 + 4 * NS) * sizeof(float);
    (void)hipFuncSetAttribute(reinterpret_cast<const void*>(scan_kernel),
                              hipFuncAttributeMaxDynamicSharedMemorySize, (int)shmem);
    scan_kernel<<<dim3(BATCH), blk, shmem, stream>>>(lg_ws, kv_ws, bt_ws, q_ws,
                                                     b_beta, out, s_final);
}
